// Encoder_LaplaceGNN_ZINCSAGE_55714315764104
// MI455X (gfx1250) — matrix operand order repaired and checked by reading, not compiled, not run on hardware
//
#include <hip/hip_runtime.h>

#define NN 65536
#define EE 524288
#define HH 256

typedef __attribute__((ext_vector_type(16))) __bf16 v16bf;
typedef __attribute__((ext_vector_type(8)))  __bf16 v8bf;
typedef __attribute__((ext_vector_type(8)))  float  v8f;

// CDNA5 async global->LDS copy of 16 bytes (no VGPR staging, tracked by ASYNCcnt).
// LDS address from the VDST VGPR; global addr in VADDR pair; no SADDR.
__device__ __forceinline__ void async_ld16(unsigned lds_off, const void* gptr) {
    asm volatile("global_load_async_to_lds_b128 %0, %1, off"
                 :: "v"(lds_off), "v"(gptr) : "memory");
}
__device__ __forceinline__ void wait_async0() {
    asm volatile("s_wait_asynccnt 0x0" ::: "memory");
}

// ---------------- degree ----------------
__global__ void k_deg(const int* __restrict__ col, float* __restrict__ deg) {
    int e = blockIdx.x * blockDim.x + threadIdx.x;
    if (e < EE) atomicAdd(&deg[col[e]], 1.0f);
}

__global__ void k_deginv(const float* __restrict__ deg, float* __restrict__ dinv) {
    int n = blockIdx.x * blockDim.x + threadIdx.x;
    if (n < NN) dinv[n] = 1.0f / fmaxf(deg[n], 1.0f);
}

// ---------------- node embedding gather ----------------
__global__ void k_embed(const int* __restrict__ x, const float* __restrict__ emb,
                        float* __restrict__ h) {
    int idx = blockIdx.x * blockDim.x + threadIdx.x;       // NN*64
    int n = idx >> 6, c = (idx & 63) << 2;
    float4 v = *(const float4*)(emb + (size_t)x[n] * HH + c);
    *(float4*)(h + (size_t)n * HH + c) = v;
}

// ---------------- edge embedding scatter: h[row[e]] += emb_edge[attr[e]] ----------------
__global__ void k_edge_scatter(const int* __restrict__ rowi, const int* __restrict__ attr,
                               const float* __restrict__ embe, float* __restrict__ h) {
    int idx = blockIdx.x * blockDim.x + threadIdx.x;       // EE*64
    int e = idx >> 6, c = (idx & 63) << 2;
    float4 v = *(const float4*)(embe + (size_t)attr[e] * HH + c);
    float* p = h + (size_t)rowi[e] * HH + c;
    atomicAdd(p + 0, v.x); atomicAdd(p + 1, v.y);
    atomicAdd(p + 2, v.z); atomicAdd(p + 3, v.w);
}

// ---------------- aggregation: agg[col[e]] += s[row[e]] (mean scaling applied later) ----------------
__global__ void k_aggregate(const int* __restrict__ rowi, const int* __restrict__ coli,
                            const float* __restrict__ s, float* __restrict__ agg) {
    int idx = blockIdx.x * blockDim.x + threadIdx.x;       // EE*64
    int e = idx >> 6, c = (idx & 63) << 2;
    float4 v = *(const float4*)(s + (size_t)rowi[e] * HH + c);
    float* p = agg + (size_t)coli[e] * HH + c;
    atomicAdd(p + 0, v.x); atomicAdd(p + 1, v.y);
    atomicAdd(p + 2, v.z); atomicAdd(p + 3, v.w);
}

// ---------------- fp32 -> bf16 weight conversion ----------------
__global__ void k_cvt_bf16(const float* __restrict__ src, __bf16* __restrict__ dst, int n) {
    int i = blockIdx.x * blockDim.x + threadIdx.x;
    if (i < n) dst[i] = (__bf16)src[i];
}

// ---------------- fused dual-GEMM via WMMA ----------------
// out[N,256] = scale(A1) @ W1^T  (+ A2 @ W2^T) (+ bias) (+ add1) (+ add2)
// W row-major [256 out][256 k] bf16; A fp32 row-major, converted to bf16 in LDS.
// B tiles are staged into LDS with CDNA5 async global->LDS copies.
__global__ __launch_bounds__(256)
void k_gemm(const float* __restrict__ A1, const float* __restrict__ rowscale,
            const __bf16* __restrict__ W1,
            const float* __restrict__ A2, const __bf16* __restrict__ W2,
            const float* __restrict__ bias,
            const float* __restrict__ add1, const float* __restrict__ add2,
            float* __restrict__ out) {
    __shared__ __bf16 lA1[128][40];   // 128 rows x 32 K (+pad, 80B row stride)
    __shared__ __bf16 lA2[128][40];
    __shared__ __bf16 lB1[64][40];    // 64 out-cols x 32 K (+pad)
    __shared__ __bf16 lB2[64][40];

    const int wave = threadIdx.x >> 5;
    const int lane = threadIdx.x & 31;
    const int m0 = blockIdx.y * 128;          // row tile
    const int n0 = blockIdx.x * 64;           // col tile
    const bool hasA2 = (A2 != nullptr);

    v8f acc[4] = {};

    for (int kk = 0; kk < 256; kk += 32) {
        __syncthreads();

        // --- B staging: async copy, 16B per thread per matrix (64x32 bf16 = 4KB) ---
        {
            int n = threadIdx.x >> 2;           // 0..63 (out-col within tile)
            int kc = (threadIdx.x & 3) << 3;    // 0,8,16,24 (bf16 elems = 16B chunks)
            async_ld16((unsigned)(uintptr_t)&lB1[n][kc],
                       W1 + (size_t)(n0 + n) * HH + kk + kc);
            if (hasA2)
                async_ld16((unsigned)(uintptr_t)&lB2[n][kc],
                           W2 + (size_t)(n0 + n) * HH + kk + kc);
        }

        // --- A staging: 128x32 fp32 -> bf16 (row-scaled for mean aggregation) ---
        for (int g = threadIdx.x; g < 1024; g += 256) {
            int r = g >> 3, c = (g & 7) << 2;
            size_t off = (size_t)(m0 + r) * HH + kk + c;
            float sc = rowscale ? rowscale[m0 + r] : 1.0f;
            float4 v = *(const float4*)(A1 + off);
            lA1[r][c + 0] = (__bf16)(v.x * sc);
            lA1[r][c + 1] = (__bf16)(v.y * sc);
            lA1[r][c + 2] = (__bf16)(v.z * sc);
            lA1[r][c + 3] = (__bf16)(v.w * sc);
            if (hasA2) {
                float4 u = *(const float4*)(A2 + off);
                lA2[r][c + 0] = (__bf16)u.x;
                lA2[r][c + 1] = (__bf16)u.y;
                lA2[r][c + 2] = (__bf16)u.z;
                lA2[r][c + 3] = (__bf16)u.w;
            }
            if (kk + 32 < 256) __builtin_prefetch(A1 + off + 32, 0, 1);
        }

        wait_async0();       // our async B chunks are in LDS
        __syncthreads();     // everyone's chunks visible

        // --- A fragment: 16x32 bf16 (lane<16: M=lane, K 0..15; lane>=16: K 16..31)
        const int ar = (wave << 4) + (lane & 15);
        const int ko = (lane >> 4) << 4;
        v8bf a1lo = *(const v8bf*)&lA1[ar][ko];
        v8bf a1hi = *(const v8bf*)&lA1[ar][ko + 8];
        v16bf af1 = __builtin_shufflevector(a1lo, a1hi,
                        0,1,2,3,4,5,6,7,8,9,10,11,12,13,14,15);
        v16bf af2;
        if (hasA2) {
            v8bf a2lo = *(const v8bf*)&lA2[ar][ko];
            v8bf a2hi = *(const v8bf*)&lA2[ar][ko + 8];
            af2 = __builtin_shufflevector(a2lo, a2hi,
                        0,1,2,3,4,5,6,7,8,9,10,11,12,13,14,15);
        }

        // --- B fragments from LDS: lane holds column t*16+(lane%16), 16 K vals at (lane/16)*16
        const int bn = lane & 15;
        const int bkl = (lane >> 4) << 4;       // 0 or 16 (local K)
        v16bf bfr1[4];
#pragma unroll
        for (int t = 0; t < 4; ++t) {
            v8bf lo = *(const v8bf*)&lB1[t * 16 + bn][bkl];
            v8bf hi = *(const v8bf*)&lB1[t * 16 + bn][bkl + 8];
            bfr1[t] = __builtin_shufflevector(lo, hi,
                        0,1,2,3,4,5,6,7,8,9,10,11,12,13,14,15);
        }
#pragma unroll
        for (int t = 0; t < 4; ++t)
            acc[t] = __builtin_amdgcn_wmma_f32_16x16x32_bf16(
                false, af1, false, bfr1[t], (short)0, acc[t], false, false);

        if (hasA2) {
            v16bf bfr2[4];
#pragma unroll
            for (int t = 0; t < 4; ++t) {
                v8bf lo = *(const v8bf*)&lB2[t * 16 + bn][bkl];
                v8bf hi = *(const v8bf*)&lB2[t * 16 + bn][bkl + 8];
                bfr2[t] = __builtin_shufflevector(lo, hi,
                            0,1,2,3,4,5,6,7,8,9,10,11,12,13,14,15);
            }
#pragma unroll
            for (int t = 0; t < 4; ++t)
                acc[t] = __builtin_amdgcn_wmma_f32_16x16x32_bf16(
                    false, af2, false, bfr2[t], (short)0, acc[t], false, false);
        }
    }

    // epilogue: C layout — VGPR i, lanes 0-15: M=i,N=lane; lanes 16-31: M=8+i,N=lane-16
    const int rbase = (wave << 4) + ((lane >> 4) << 3);
#pragma unroll
    for (int t = 0; t < 4; ++t) {
        const int colg = n0 + t * 16 + (lane & 15);
        const float bv = bias ? bias[colg] : 0.0f;
#pragma unroll
        for (int i = 0; i < 8; ++i) {
            size_t off = (size_t)(m0 + rbase + i) * HH + colg;
            float v = acc[t][i] + bv;
            if (add1) v += add1[off];
            if (add2) v += add2[off];
            out[off] = v;
        }
    }
}

// ---------------- BatchNorm stats: per-column sum / sumsq ----------------
__global__ void k_bn_stats(const float* __restrict__ z, float* __restrict__ stats) {
    int c = threadIdx.x;                      // 256 columns
    int r0 = blockIdx.x * 256;                // 256 blocks x 256 rows
    float s = 0.f, ss = 0.f;
    for (int r = 0; r < 256; ++r) {
        float v = z[(size_t)(r0 + r) * HH + c];
        s += v; ss += v * v;
    }
    atomicAdd(&stats[c], s);
    atomicAdd(&stats[HH + c], ss);
}

// ---------------- BN affine + PReLU ----------------
__global__ void k_bn_apply(const float* __restrict__ z, const float* __restrict__ stats,
                           const float* __restrict__ g, const float* __restrict__ b,
                           const float* __restrict__ a, float* __restrict__ out) {
    int idx = blockIdx.x * blockDim.x + threadIdx.x;       // NN*64
    int c4 = (idx & 63) << 2;
    size_t off = ((size_t)(idx >> 6) * HH) + c4;
    const float inv_n = 1.0f / (float)NN;
    const float alpha = a[0];
    float4 v = *(const float4*)(z + off);
    float r[4] = {v.x, v.y, v.z, v.w};
#pragma unroll
    for (int i = 0; i < 4; ++i) {
        int c = c4 + i;
        float m  = stats[c] * inv_n;
        float var = stats[HH + c] * inv_n - m * m;
        float y = (r[i] - m) * rsqrtf(var + 1e-5f) * g[c] + b[c];
        r[i] = (y >= 0.f) ? y : alpha * y;
    }
    float4 o = {r[0], r[1], r[2], r[3]};
    *(float4*)(out + off) = o;
}

// ---------------- host-side orchestration ----------------
extern "C" void kernel_launch(void* const* d_in, const int* in_sizes, int n_in,
                              void* d_out, int out_size, void* d_ws, size_t ws_size,
                              hipStream_t stream) {
    const int* x      = (const int*)d_in[0];
    const int* eidx   = (const int*)d_in[1];
    const int* row    = eidx;
    const int* col    = eidx + EE;
    const int* eattr  = (const int*)d_in[2];
    const float* emb_node = (const float*)d_in[3];
    const float* emb_edge = (const float*)d_in[4];
    const float* Wl0 = (const float*)d_in[5];
    const float* bl0 = (const float*)d_in[6];
    const float* Wr0 = (const float*)d_in[7];
    const float* Wl1 = (const float*)d_in[8];
    const float* bl1 = (const float*)d_in[9];
    const float* Wr1 = (const float*)d_in[10];
    const float* Wl2 = (const float*)d_in[11];
    const float* bl2 = (const float*)d_in[12];
    const float* Wr2 = (const float*)d_in[13];
    const float* Wsk0 = (const float*)d_in[14];
    const float* Wsk1 = (const float*)d_in[15];
    const float* g0 = (const float*)d_in[16];
    const float* b0 = (const float*)d_in[17];
    const float* g1 = (const float*)d_in[18];
    const float* b1 = (const float*)d_in[19];
    const float* g2 = (const float*)d_in[20];
    const float* b2 = (const float*)d_in[21];
    const float* a0 = (const float*)d_in[22];
    const float* a1 = (const float*)d_in[23];
    const float* a2 = (const float*)d_in[24];

    char* w = (char*)d_ws;
    auto carve = [&](size_t bytes) {
        void* p = (void*)w;
        w += (bytes + 255) & ~(size_t)255;
        return p;
    };
    float* h0   = (float*)carve((size_t)NN * HH * 4);
    float* h1   = (float*)carve((size_t)NN * HH * 4);
    float* h2   = (float*)carve((size_t)NN * HH * 4);
    float* sbuf = (float*)carve((size_t)NN * HH * 4);
    float* agg  = (float*)carve((size_t)NN * HH * 4);
    float* deg  = (float*)carve((size_t)NN * 4);
    float* dinv = (float*)carve((size_t)NN * 4);
    float* stats = (float*)carve(2 * HH * 4);
    __bf16* wbWl0 = (__bf16*)carve((size_t)HH * HH * 2);
    __bf16* wbWr0 = (__bf16*)carve((size_t)HH * HH * 2);
    __bf16* wbWl1 = (__bf16*)carve((size_t)HH * HH * 2);
    __bf16* wbWr1 = (__bf16*)carve((size_t)HH * HH * 2);
    __bf16* wbWl2 = (__bf16*)carve((size_t)HH * HH * 2);
    __bf16* wbWr2 = (__bf16*)carve((size_t)HH * HH * 2);
    __bf16* wbSk0 = (__bf16*)carve((size_t)HH * HH * 2);
    __bf16* wbSk1 = (__bf16*)carve((size_t)HH * HH * 2);

    float* zbuf = (float*)d_out;   // reuse output buffer as pre-BN scratch each layer

    const dim3 B256(256);
    const dim3 Gdeg(EE / 256), Gn(NN / 256);
    const dim3 Gnode(NN * 64 / 256), Gedge(EE * 64 / 256);
    const dim3 Gw(HH * HH / 256);
    const dim3 Ggemm(4, NN / 128);   // 4 col tiles x 512 row tiles

    // degrees
    hipMemsetAsync(deg, 0, (size_t)NN * 4, stream);
    k_deg<<<Gdeg, B256, 0, stream>>>(col, deg);
    k_deginv<<<Gn, B256, 0, stream>>>(deg, dinv);

    // h0 = emb_node[x]; h0[row] += emb_edge[attr]
    k_embed<<<Gnode, B256, 0, stream>>>(x, emb_node, h0);
    k_edge_scatter<<<Gedge, B256, 0, stream>>>(row, eattr, emb_edge, h0);

    // weight conversions (fp32 -> bf16)
    k_cvt_bf16<<<Gw, B256, 0, stream>>>(Wl0, wbWl0, HH * HH);
    k_cvt_bf16<<<Gw, B256, 0, stream>>>(Wr0, wbWr0, HH * HH);
    k_cvt_bf16<<<Gw, B256, 0, stream>>>(Wl1, wbWl1, HH * HH);
    k_cvt_bf16<<<Gw, B256, 0, stream>>>(Wr1, wbWr1, HH * HH);
    k_cvt_bf16<<<Gw, B256, 0, stream>>>(Wl2, wbWl2, HH * HH);
    k_cvt_bf16<<<Gw, B256, 0, stream>>>(Wr2, wbWr2, HH * HH);
    k_cvt_bf16<<<Gw, B256, 0, stream>>>(Wsk0, wbSk0, HH * HH);
    k_cvt_bf16<<<Gw, B256, 0, stream>>>(Wsk1, wbSk1, HH * HH);

    // ---- layer 0: z0 = mean-agg(h0)@Wl0^T + bl0 + h0@Wr0^T ; h1 = prelu(bn(z0))
    hipMemsetAsync(agg, 0, (size_t)NN * HH * 4, stream);
    k_aggregate<<<Gedge, B256, 0, stream>>>(row, col, h0, agg);
    k_gemm<<<Ggemm, B256, 0, stream>>>(agg, dinv, wbWl0, h0, wbWr0, bl0,
                                       nullptr, nullptr, zbuf);
    hipMemsetAsync(stats, 0, 2 * HH * 4, stream);
    k_bn_stats<<<dim3(NN / 256), B256, 0, stream>>>(zbuf, stats);
    k_bn_apply<<<Gnode, B256, 0, stream>>>(zbuf, stats, g0, b0, a0, h1);

    // ---- layer 1: s1 = h1 + h0@Wskip0^T
    k_gemm<<<Ggemm, B256, 0, stream>>>(h0, nullptr, wbSk0, nullptr, nullptr, nullptr,
                                       h1, nullptr, sbuf);
    hipMemsetAsync(agg, 0, (size_t)NN * HH * 4, stream);
    k_aggregate<<<Gedge, B256, 0, stream>>>(row, col, sbuf, agg);
    k_gemm<<<Ggemm, B256, 0, stream>>>(agg, dinv, wbWl1, sbuf, wbWr1, bl1,
                                       nullptr, nullptr, zbuf);
    hipMemsetAsync(stats, 0, 2 * HH * 4, stream);
    k_bn_stats<<<dim3(NN / 256), B256, 0, stream>>>(zbuf, stats);
    k_bn_apply<<<Gnode, B256, 0, stream>>>(zbuf, stats, g1, b1, a1, h2);

    // ---- layer 2: s2 = h1 + h2 + h0@Wskip1^T
    k_gemm<<<Ggemm, B256, 0, stream>>>(h0, nullptr, wbSk1, nullptr, nullptr, nullptr,
                                       h1, h2, sbuf);
    hipMemsetAsync(agg, 0, (size_t)NN * HH * 4, stream);
    k_aggregate<<<Gedge, B256, 0, stream>>>(row, col, sbuf, agg);
    k_gemm<<<Ggemm, B256, 0, stream>>>(agg, dinv, wbWl2, sbuf, wbWr2, bl2,
                                       nullptr, nullptr, zbuf);
    hipMemsetAsync(stats, 0, 2 * HH * 4, stream);
    k_bn_stats<<<dim3(NN / 256), B256, 0, stream>>>(zbuf, stats);
    k_bn_apply<<<Gnode, B256, 0, stream>>>(zbuf, stats, g2, b2, a2, (float*)d_out);
}
